// MultiHeadSelfAttention_69853348102594
// MI455X (gfx1250) — matrix operand order repaired and checked by reading, not compiled, not run on hardware
//
#include <hip/hip_runtime.h>

// ---------------------------------------------------------------------------
// MHA forward for MI455X (gfx1250), wave32, bf16 WMMA with f32 accumulation.
//   B=2, L=2048, D=1024, H=16, hd=64
// Round 3: fix tensor_load_to_lds arity (6-arg toolchain variant).
// ---------------------------------------------------------------------------

typedef __bf16 bf16;
typedef __attribute__((ext_vector_type(16))) __bf16 v16bf;
typedef __attribute__((ext_vector_type(8)))  __bf16 v8bf;
typedef __attribute__((ext_vector_type(4)))  __bf16 v4bf;
typedef __attribute__((ext_vector_type(8)))  float  v8f;
typedef __attribute__((ext_vector_type(4)))  float  v4f;
typedef __attribute__((ext_vector_type(4)))  unsigned int u32x4;
typedef __attribute__((ext_vector_type(8)))  int i32x8;
typedef __attribute__((ext_vector_type(4)))  int i32x4;

#define D_MODEL   1024
#define NUM_HEADS 16
#define HEAD_DIM  64
#define SEQ_LEN   2048
#define BATCH     2
#define M_ROWS    (BATCH * SEQ_LEN)   // 4096

#if __has_builtin(__builtin_amdgcn_tensor_load_to_lds) && \
    __has_builtin(__builtin_amdgcn_s_wait_tensorcnt)
#define USE_TDM 1
#else
#define USE_TDM 0
#endif

__device__ __forceinline__ v16bf join16(v8bf lo, v8bf hi) {
    v16bf r;
#pragma unroll
    for (int i = 0; i < 8; ++i) { r[i] = lo[i]; r[i + 8] = hi[i]; }
    return r;
}

__device__ __forceinline__ v8f wmma_bf16(v16bf a, v16bf b, v8f c) {
    // D = A(16x32 bf16) * B(32x16 bf16) + C(16x16 f32)
    return __builtin_amdgcn_wmma_f32_16x16x32_bf16(
        /*neg_a=*/false, a, /*neg_b=*/false, b,
        /*c_mod=*/(short)0, c, /*reuse_a=*/false, /*reuse_b=*/false);
}

// ---------------------------------------------------------------------------
// Kernel 0: f32 -> bf16 conversion (vectorized x4)
// ---------------------------------------------------------------------------
__global__ __launch_bounds__(256) void cvt_f32_bf16(const v4f* __restrict__ src,
                                                    v4bf* __restrict__ dst, int n4) {
    int i = blockIdx.x * 256 + threadIdx.x;
    if (i < n4) {
        v4f x = src[i];
        v4bf y;
#pragma unroll
        for (int j = 0; j < 4; ++j) y[j] = (bf16)x[j];
        dst[i] = y;
    }
}

// ---------------------------------------------------------------------------
// Software-pipelined 16(M) x 64(N) x 1024(K) wave-level GEMM core.
// Two register stages: loads for k-step kk+32 are issued before the WMMAs
// consuming k-step kk, so vmem latency overlaps the matrix pipe.
//   arowp = &A[row(lane)][0]  (row-major, stride D_MODEL)
//   wbase = &W[n_tile_base][0] (row-major, stride D_MODEL; B = W^T)
// ---------------------------------------------------------------------------
__device__ __forceinline__ void gemm16x64_pipe(const bf16* __restrict__ arowp,
                                               const bf16* __restrict__ wbase,
                                               int l16, int half, v8f acc[4]) {
    v8bf a_lo[2], a_hi[2];
    v8bf b_lo[2][4], b_hi[2][4];

    auto LOAD = [&](int st, int kk) {
        const int ak = kk + half * 8;
        a_lo[st] = *(const v8bf*)(arowp + ak);
        a_hi[st] = *(const v8bf*)(arowp + ak + 16);
        const int bk = kk + half * 8;
#pragma unroll
        for (int t = 0; t < 4; ++t) {
            const bf16* wrow = wbase + (size_t)(t * 16 + l16) * D_MODEL + bk;
            b_lo[st][t] = *(const v8bf*)wrow;
            b_hi[st][t] = *(const v8bf*)(wrow + 16);
        }
    };
    auto COMP = [&](int st) {
        v16bf af = join16(a_lo[st], a_hi[st]);
#pragma unroll
        for (int t = 0; t < 4; ++t)
            acc[t] = wmma_bf16(af, join16(b_lo[st][t], b_hi[st][t]), acc[t]);
    };

    LOAD(0, 0);
    for (int kk = 0; kk < D_MODEL - 64; kk += 64) {
        LOAD(1, kk + 32);
        COMP(0);
        LOAD(0, kk + 64);
        COMP(1);
    }
    LOAD(1, D_MODEL - 32);
    COMP(0);
    COMP(1);
}

// ---------------------------------------------------------------------------
// Kernel 1: fused QKV projection. out = x @ W^T + b -> [B,H,L,hd] bf16.
// Block = 8 waves; wave computes 16 x 64; grid = (3072/64, 4096/128).
// ---------------------------------------------------------------------------
__global__ __launch_bounds__(256)
void qkv_proj_kernel(const bf16* __restrict__ xb,
                     const bf16* __restrict__ wqb, const bf16* __restrict__ wkb,
                     const bf16* __restrict__ wvb,
                     const float* __restrict__ bq, const float* __restrict__ bk,
                     const float* __restrict__ bv,
                     bf16* __restrict__ qo, bf16* __restrict__ ko, bf16* __restrict__ vo) {
    const int lane = threadIdx.x & 31;
    const int wave = threadIdx.x >> 5;
    const int half = lane >> 4;
    const int l16  = lane & 15;

    const int m0 = blockIdx.y * 128 + wave * 16;
    const int n0 = blockIdx.x * 64;                 // [0, 3072)
    const int wsel = n0 >> 10;                      // 0=q 1=k 2=v
    const int nin0 = n0 & 1023;

    const bf16*  W    = (wsel == 0) ? wqb : (wsel == 1) ? wkb : wvb;
    const float* bias = (wsel == 0) ? bq  : (wsel == 1) ? bk  : bv;
    bf16*        outp = (wsel == 0) ? qo  : (wsel == 1) ? ko  : vo;

    v8f acc[4] = {};
    gemm16x64_pipe(xb + (size_t)(m0 + l16) * D_MODEL,
                   W + (size_t)nin0 * D_MODEL, l16, half, acc);

    // Epilogue: bias + scatter to [B, H, L, hd]
#pragma unroll
    for (int t = 0; t < 4; ++t) {
        const int nin = nin0 + t * 16 + l16;
        const float bval = bias[nin];
        const int h = nin >> 6, d = nin & 63;
#pragma unroll
        for (int r = 0; r < 8; ++r) {
            const int m = m0 + half * 8 + r;
            const int b = m >> 11, l = m & 2047;
            const size_t idx = (((size_t)b * NUM_HEADS + h) * SEQ_LEN + l) * HEAD_DIM + d;
            outp[idx] = (bf16)(acc[t][r] + bval);
        }
    }
}

// ---------------------------------------------------------------------------
// TDM helper: issue a 2D tile load (32 rows x 64 cols of bf16) from a
// row-major [SEQ_LEN x HEAD_DIM] tensor into LDS at lds_byte_off.
// D# layout per CDNA5 ISA (groups 2/3 zero => 2D).
// This toolchain exposes the 6-arg builtin form (extra zero-filled group).
// ---------------------------------------------------------------------------
#if USE_TDM
__device__ __forceinline__ void tdm_load_k_tile(const bf16* tile_src,
                                                unsigned lds_byte_off) {
    const unsigned long long ga = (unsigned long long)(uintptr_t)tile_src;
    u32x4 g0;
    g0[0] = 1u;                                   // count=1 (valid), user mode
    g0[1] = lds_byte_off;                         // lds_addr (bytes)
    g0[2] = (unsigned)ga;                         // global_addr[31:0]
    g0[3] = (unsigned)((ga >> 32) & 0x1FFFFFFu)   // global_addr[56:32]
            | (2u << 30);                         // type = 2 ("image")
    i32x8 g1;
    g1[0] = 1 << 16;                              // data_size = 1 -> 2 bytes
    g1[1] = HEAD_DIM << 16;                       // tensor_dim0 = 64
    g1[2] = SEQ_LEN << 16;                        // tensor_dim1 = 2048
    g1[3] = HEAD_DIM << 16;                       // tile_dim0 = 64
    g1[4] = 32;                                   // tile_dim1 = 32
    g1[5] = HEAD_DIM;                             // tensor_dim0_stride = 64
    g1[6] = 0;
    g1[7] = 0;
    const i32x4 z4 = {0, 0, 0, 0};
    const i32x8 z8 = {0, 0, 0, 0, 0, 0, 0, 0};
    __builtin_amdgcn_tensor_load_to_lds(g0, g1, z4, z4, z8, /*cpol=*/0);
}
#endif

// ---------------------------------------------------------------------------
// Kernel 2: flash attention per (b,h). Block = 8 waves, covers 128 q-rows.
// Double-buffered K/V^T tiles in dynamic LDS:
//   - K tile staged by the Tensor Data Mover (wave 0 issues; TENSORcnt sync),
//     prefetching the next tile while the current one is computed.
//   - V tile held in registers (prefetched) and scatter-transposed to LDS.
// Per 32-key step: 4 WMMAs for scores, online softmax (shfl_xor over 16-lane
// halves), P -> per-wave LDS tile (C-layout -> A-layout), 4 WMMAs for O.
// Dynamic LDS: kbuf[2][32][64] @ 0, vtbuf[2][64][32] @ 8192 (16 KB total).
// ---------------------------------------------------------------------------
__global__ __launch_bounds__(256)
void attention_kernel(const bf16* __restrict__ q, const bf16* __restrict__ k,
                      const bf16* __restrict__ v, bf16* __restrict__ ctx) {
    extern __shared__ __align__(16) char smem[];
    bf16 (*kbuf)[32][64]  = (bf16 (*)[32][64])(smem);
    bf16 (*vtbuf)[64][32] = (bf16 (*)[64][32])(smem + 8192);
    __shared__ __align__(16) bf16 pbuf[8][16][32];   // per-wave P tile (static)

    const int lane = threadIdx.x & 31;
    const int wave = threadIdx.x >> 5;
    const int half = lane >> 4;
    const int l16  = lane & 15;

    const int bh = blockIdx.x;                       // b*16 + h
    const int q0 = blockIdx.y * 128 + wave * 16;

    const bf16* qp = q + ((size_t)bh * SEQ_LEN + q0) * HEAD_DIM;
    const bf16* kp = k + (size_t)bh * SEQ_LEN * HEAD_DIM;
    const bf16* vp = v + (size_t)bh * SEQ_LEN * HEAD_DIM;

    // Q fragments (hd = 64 -> two 16x32 A fragments), kept in registers.
    v16bf qf[2];
    {
        const bf16* qr = qp + (size_t)l16 * HEAD_DIM;
#pragma unroll
        for (int j = 0; j < 2; ++j) {
            const int ko = j * 32 + half * 8;
            qf[j] = join16(*(const v8bf*)(qr + ko), *(const v8bf*)(qr + ko + 16));
        }
    }

    v8f oacc[4] = {};
    float mrow[8], lrow[8];
#pragma unroll
    for (int r = 0; r < 8; ++r) { mrow[r] = -1e30f; lrow[r] = 0.0f; }

    // cooperative-load geometry: each thread owns (key, 8-elem chunk)
    const int ckey = threadIdx.x >> 3;               // 0..31
    const int cch  = (threadIdx.x & 7) * 8;          // 0..56

#if USE_TDM
    const unsigned dynbase = (unsigned)__builtin_amdgcn_groupstaticsize();
#endif

    // ---- prologue: stage tile 0 ----
    v8bf vreg = *(const v8bf*)(vp + (size_t)ckey * HEAD_DIM + cch);
#if USE_TDM
    if (wave == 0) tdm_load_k_tile(kp, dynbase + 0 * 4096);
#else
    {
        v8bf kvv = *(const v8bf*)(kp + (size_t)ckey * HEAD_DIM + cch);
        *(v8bf*)(&kbuf[0][ckey][cch]) = kvv;
    }
#endif

    for (int j0 = 0; j0 < SEQ_LEN; j0 += 32) {
        const int buf = (j0 >> 5) & 1;
        const bool hasnext = (j0 + 32) < SEQ_LEN;

        // scatter current V tile (registers -> transposed LDS)
#pragma unroll
        for (int e = 0; e < 8; ++e) vtbuf[buf][cch + e][ckey] = vreg[e];

        // prefetch next tile: V into registers, K via TDM (or plain loads)
        v8bf vnext;
        if (hasnext) {
            vnext = *(const v8bf*)(vp + (size_t)(j0 + 32 + ckey) * HEAD_DIM + cch);
#if USE_TDM
            if (wave == 0)
                tdm_load_k_tile(kp + (size_t)(j0 + 32) * HEAD_DIM,
                                dynbase + (buf ^ 1) * 4096);
#else
            v8bf kvv = *(const v8bf*)(kp + (size_t)(j0 + 32 + ckey) * HEAD_DIM + cch);
            *(v8bf*)(&kbuf[buf ^ 1][ckey][cch]) = kvv;
#endif
        }
#if USE_TDM
        if (wave == 0) {
            if (hasnext) __builtin_amdgcn_s_wait_tensorcnt(1);  // current tile done
            else         __builtin_amdgcn_s_wait_tensorcnt(0);
        }
#endif
        __syncthreads();

        // --- scores: S[16q x 32keys] as two 16x16 f32 tiles (4 WMMAs) ---
        v8f s[2];
#pragma unroll
        for (int t = 0; t < 2; ++t) {
            v8f sa = {};
            const int ck = t * 16 + l16;             // key column (B-frag lane)
#pragma unroll
            for (int st = 0; st < 2; ++st) {
                const bf16* kr = &kbuf[buf][ck][st * 32 + half * 8];
                v16bf bfrag = join16(*(const v8bf*)kr, *(const v8bf*)(kr + 16));
                sa = wmma_bf16(qf[st], bfrag, sa);
            }
            s[t] = sa;
        }

        // --- online softmax over the 32 new keys ---
#pragma unroll
        for (int r = 0; r < 8; ++r) {
            float s0 = s[0][r] * 0.125f;             // 1/sqrt(64)
            float s1 = s[1][r] * 0.125f;
            float mx = fmaxf(s0, s1);
#pragma unroll
            for (int off = 1; off < 16; off <<= 1)
                mx = fmaxf(mx, __shfl_xor(mx, off, 32));
            const float mnew = fmaxf(mrow[r], mx);
            const float p0 = __expf(s0 - mnew);
            const float p1 = __expf(s1 - mnew);
            float psum = p0 + p1;
#pragma unroll
            for (int off = 1; off < 16; off <<= 1)
                psum += __shfl_xor(psum, off, 32);
            const float scale = __expf(mrow[r] - mnew);
            lrow[r] = lrow[r] * scale + psum;
            mrow[r] = mnew;
#pragma unroll
            for (int c = 0; c < 4; ++c) oacc[c][r] *= scale;
            const int row = half * 8 + r;            // P: row-major q x 32keys
            pbuf[wave][row][l16]      = (bf16)p0;
            pbuf[wave][row][16 + l16] = (bf16)p1;
        }

        // --- P back as 16x32 A fragment (C-layout -> A-layout via LDS) ---
        v16bf pf;
        {
            const bf16* pr = &pbuf[wave][l16][0];
            const int ko = half * 8;
            pf = join16(*(const v8bf*)(pr + ko), *(const v8bf*)(pr + ko + 16));
        }

        // --- O += P @ V : 4 WMMAs over the four 16-wide d tiles ---
#pragma unroll
        for (int c = 0; c < 4; ++c) {
            const bf16* vr = &vtbuf[buf][c * 16 + l16][half * 8];
            v16bf vfrag = join16(*(const v8bf*)vr, *(const v8bf*)(vr + 16));
            oacc[c] = wmma_bf16(pf, vfrag, oacc[c]);
        }
        __syncthreads();

        if (hasnext) vreg = vnext;
    }

    // --- epilogue: normalize, store ctx as [B, L, D] bf16, D = h*64 + d ---
    const int b = bh >> 4, h = bh & 15;
#pragma unroll
    for (int r = 0; r < 8; ++r) {
        const float inv = 1.0f / lrow[r];
        const int qrow = q0 + half * 8 + r;
        const size_t base = ((size_t)b * SEQ_LEN + qrow) * D_MODEL + h * HEAD_DIM;
#pragma unroll
        for (int c = 0; c < 4; ++c)
            ctx[base + c * 16 + l16] = (bf16)(oacc[c][r] * inv);
    }
}

// ---------------------------------------------------------------------------
// Kernel 3: output projection.  out = ctx @ Wo^T + bo  (f32 output)
// ---------------------------------------------------------------------------
__global__ __launch_bounds__(256)
void out_proj_kernel(const bf16* __restrict__ ctxb, const bf16* __restrict__ wob,
                     const float* __restrict__ bo, float* __restrict__ out) {
    const int lane = threadIdx.x & 31;
    const int wave = threadIdx.x >> 5;
    const int half = lane >> 4;
    const int l16  = lane & 15;

    const int m0 = blockIdx.y * 128 + wave * 16;
    const int n0 = blockIdx.x * 64;

    v8f acc[4] = {};
    gemm16x64_pipe(ctxb + (size_t)(m0 + l16) * D_MODEL,
                   wob + (size_t)n0 * D_MODEL, l16, half, acc);

#pragma unroll
    for (int t = 0; t < 4; ++t) {
        const int n = n0 + t * 16 + l16;
        const float bval = bo[n];
#pragma unroll
        for (int r = 0; r < 8; ++r) {
            const int m = m0 + half * 8 + r;
            out[(size_t)m * D_MODEL + n] = acc[t][r] + bval;
        }
    }
}

// ---------------------------------------------------------------------------
// Host-side launch
// ---------------------------------------------------------------------------
extern "C" void kernel_launch(void* const* d_in, const int* in_sizes, int n_in,
                              void* d_out, int out_size, void* d_ws, size_t ws_size,
                              hipStream_t stream) {
    (void)in_sizes; (void)n_in; (void)out_size; (void)ws_size;

    const float* x  = (const float*)d_in[0];
    const float* Wq = (const float*)d_in[1];
    const float* bq = (const float*)d_in[2];
    const float* Wk = (const float*)d_in[3];
    const float* bk = (const float*)d_in[4];
    const float* Wv = (const float*)d_in[5];
    const float* bv = (const float*)d_in[6];
    const float* Wo = (const float*)d_in[7];
    const float* bo = (const float*)d_in[8];
    float* out = (float*)d_out;

    // Workspace layout (bf16), ~50 MB total.
    char* ws = (char*)d_ws;
    const size_t xBytes = (size_t)M_ROWS * D_MODEL * sizeof(bf16);    // 8 MB
    const size_t wBytes = (size_t)D_MODEL * D_MODEL * sizeof(bf16);   // 2 MB
    const size_t hBytes = (size_t)BATCH * NUM_HEADS * SEQ_LEN * HEAD_DIM * sizeof(bf16);

    bf16* xb  = (bf16*)ws;            ws += xBytes;
    bf16* wqb = (bf16*)ws;            ws += wBytes;
    bf16* wkb = (bf16*)ws;            ws += wBytes;
    bf16* wvb = (bf16*)ws;            ws += wBytes;
    bf16* wob = (bf16*)ws;            ws += wBytes;
    bf16* qb  = (bf16*)ws;            ws += hBytes;
    bf16* kb  = (bf16*)ws;            ws += hBytes;
    bf16* vb  = (bf16*)ws;            ws += hBytes;
    bf16* ctx = (bf16*)ws;            ws += xBytes;

    // 0) f32 -> bf16 conversions
    {
        const int nx4 = M_ROWS * D_MODEL / 4;
        cvt_f32_bf16<<<(nx4 + 255) / 256, 256, 0, stream>>>((const v4f*)x, (v4bf*)xb, nx4);
        const int nw4 = D_MODEL * D_MODEL / 4;
        cvt_f32_bf16<<<(nw4 + 255) / 256, 256, 0, stream>>>((const v4f*)Wq, (v4bf*)wqb, nw4);
        cvt_f32_bf16<<<(nw4 + 255) / 256, 256, 0, stream>>>((const v4f*)Wk, (v4bf*)wkb, nw4);
        cvt_f32_bf16<<<(nw4 + 255) / 256, 256, 0, stream>>>((const v4f*)Wv, (v4bf*)wvb, nw4);
        cvt_f32_bf16<<<(nw4 + 255) / 256, 256, 0, stream>>>((const v4f*)Wo, (v4bf*)wob, nw4);
    }

    // 1) fused QKV projection: grid = (3072/64, 4096/128)
    {
        dim3 grid(48, 32);
        qkv_proj_kernel<<<grid, 256, 0, stream>>>(xb, wqb, wkb, wvb,
                                                  bq, bk, bv, qb, kb, vb);
    }

    // 2) flash attention: grid = (B*H, L/128), 16 KB dynamic LDS
    {
        dim3 grid(BATCH * NUM_HEADS, SEQ_LEN / 128);
        attention_kernel<<<grid, 256, 16384, stream>>>(qb, kb, vb, ctx);
    }

    // 3) output projection: grid = (1024/64, 4096/128)
    {
        dim3 grid(16, 32);
        out_proj_kernel<<<grid, 256, 0, stream>>>(ctx, wob, bo, out);
    }
}
